// MultiHeadAttentionSelfManual_38671885534057
// MI455X (gfx1250) — compile-verified, matrix operand order checked by reading, not run on hardware
//
#include <hip/hip_runtime.h>

// ---------------------------------------------------------------------------
// MI455X (gfx1250) Whisper-style MHA.
// Matmuls via V_WMMA_F32_16X16X32_BF16 (wave32, f32 accumulate).
// Tile staging via GLOBAL_LOAD_ASYNC_TO_LDS_B128 (ASYNCcnt double-buffering).
//
// Fragment layouts per CDNA5 ISA 7.12.2:
//   A (16x32 bf16): lane L -> row m = L&15, K-half = (L>>4)*8;
//                   dword j holds K = {j<4 ? kh+2j : 16+kh+2(j-4)}, +1
//   B (32x16 bf16): lane L -> col n = L&15, same K pattern (K-major per lane)
//   C/D (16x16 f32): lane L -> col n = L&15; dword j -> row m = j + 8*(L>>4)
// ---------------------------------------------------------------------------

typedef __attribute__((ext_vector_type(16))) __bf16 v16bf;
typedef __attribute__((ext_vector_type(8)))  float  v8f;

union FragB { v16bf v; unsigned u[8]; };

__device__ __forceinline__ int fk(int j, int hi8) {
  return (j < 4) ? (hi8 + 2 * j) : (16 + hi8 + 2 * (j - 4));
}

__device__ __forceinline__ unsigned pack2bf(float a, float b) {
  union { __bf16 h; unsigned short s; } ca, cb;
  ca.h = (__bf16)a; cb.h = (__bf16)b;
  return (unsigned)ca.s | ((unsigned)cb.s << 16);
}

__device__ __forceinline__ v8f vzero8() {
  v8f z = {0.f, 0.f, 0.f, 0.f, 0.f, 0.f, 0.f, 0.f};
  return z;
}

// LDS byte offset: flat LDS-aperture addresses carry the offset in addr[31:0]
__device__ __forceinline__ unsigned lds_off(const void* p) {
  return (unsigned)(unsigned long long)p;
}

// Async 16B/lane copy global -> LDS (tracked by ASYNCcnt)
__device__ __forceinline__ void async_b128(const void* g, unsigned lds) {
  asm volatile("global_load_async_to_lds_b128 %0, %1, off"
               :: "v"(lds), "v"(g) : "memory");
}
#define WAIT_ASYNC(n) asm volatile("s_wait_asynccnt " #n ::: "memory")

// ---------------------------------------------------------------------------
// f32 -> bf16 elementwise (x)
// ---------------------------------------------------------------------------
__global__ void mha_cvt_bf16_kernel(const float* __restrict__ in,
                                    __bf16* __restrict__ out, int n) {
  int i = blockIdx.x * blockDim.x + threadIdx.x;
  if (i < n) out[i] = (__bf16)in[i];
}

// ---------------------------------------------------------------------------
// f32 [K][N] -> bf16 transposed [N][K] (weights; done once, reused 28x)
// ---------------------------------------------------------------------------
__global__ __launch_bounds__(256) void mha_cvt_t_kernel(
    const float* __restrict__ in, __bf16* __restrict__ out, int K, int N) {
  __shared__ float tile[32][33];
  const int tx = threadIdx.x, ty = threadIdx.y;
  const int n0 = blockIdx.x * 32, k0 = blockIdx.y * 32;
#pragma unroll
  for (int i = 0; i < 32; i += 8)
    tile[ty + i][tx] = in[(size_t)(k0 + ty + i) * N + n0 + tx];
  __syncthreads();
#pragma unroll
  for (int i = 0; i < 32; i += 8)
    out[(size_t)(n0 + ty + i) * K + k0 + tx] = (__bf16)tile[tx][ty + i];
}

// ---------------------------------------------------------------------------
// bf16 GEMM: C[M,N] = A[M,K] @ BT[N,K]^T (+bias). 256 thr = 8 waves.
// Block tile 128x128, wave tile 32x64, K-step 32, async double-buffered LDS.
// Optional outputs: f32 (+bias), bf16*scale, bf16*scale transposed [N][M].
// ---------------------------------------------------------------------------
__global__ __launch_bounds__(256) void mha_gemm_bf16_kernel(
    const __bf16* __restrict__ A, const __bf16* __restrict__ BT,
    const float* __restrict__ bias, float* __restrict__ outF,
    __bf16* __restrict__ outB, __bf16* __restrict__ outBT,
    float scaleB, int M, int N, int K) {
  __shared__ __align__(16) __bf16 As[2][128][32];   // 16 KB
  __shared__ __align__(16) __bf16 Bs[2][128][32];   // 16 KB (rows of BT)

  const int tid  = threadIdx.x;
  const int wv   = tid >> 5, lane = tid & 31;
  const int ln   = lane & 15, hi8 = (lane >> 4) * 8;
  const int wm   = wv >> 1,  wn   = wv & 1;
  const int m0   = blockIdx.y * 128, n0 = blockIdx.x * 128;

  v8f acc[2][4];
#pragma unroll
  for (int a = 0; a < 2; ++a)
#pragma unroll
    for (int b = 0; b < 4; ++b) acc[a][b] = vzero8();

  // async staging map: chunk c -> row = wv*16 + c*8 + lane/4, 16B at col (lane&3)*8
  const int se = (lane & 3) * 8;
  auto stage = [&](int kk, int buf) {
#pragma unroll
    for (int c = 0; c < 2; ++c) {
      const int r = wv * 16 + c * 8 + (lane >> 2);
      async_b128(A  + (size_t)(m0 + r) * K + kk + se, lds_off(&As[buf][r][se]));
      async_b128(BT + (size_t)(n0 + r) * K + kk + se, lds_off(&Bs[buf][r][se]));
    }
  };

  stage(0, 0);
  for (int kk = 0, it = 0; kk < K; kk += 32, ++it) {
    const int buf = it & 1;
    __syncthreads();                       // all waves done reading buf^1
    if (kk + 32 < K) { stage(kk + 32, buf ^ 1); WAIT_ASYNC(4); }
    else             { WAIT_ASYNC(0); }
    __syncthreads();                       // tile `it` fully resident

    FragB af[2], bf[4];
#pragma unroll
    for (int t = 0; t < 2; ++t) {
      const int r = wm * 32 + t * 16 + ln;
#pragma unroll
      for (int j = 0; j < 8; ++j)
        af[t].u[j] = *(const unsigned*)&As[buf][r][fk(j, hi8)];
    }
#pragma unroll
    for (int t = 0; t < 4; ++t) {
      const int c = wn * 64 + t * 16 + ln;
#pragma unroll
      for (int j = 0; j < 8; ++j)
        bf[t].u[j] = *(const unsigned*)&Bs[buf][c][fk(j, hi8)];
    }
#pragma unroll
    for (int a = 0; a < 2; ++a)
#pragma unroll
      for (int b = 0; b < 4; ++b)
        acc[a][b] = __builtin_amdgcn_wmma_f32_16x16x32_bf16(
            false, af[a].v, false, bf[b].v, (short)0, acc[a][b], false, false);
  }

  // ---- epilogue: add bias in-register, then uniform-branch store paths ----
#pragma unroll
  for (int b = 0; b < 4; ++b) {
    const float bia = bias ? bias[n0 + wn * 64 + b * 16 + ln] : 0.f;
#pragma unroll
    for (int a = 0; a < 2; ++a)
#pragma unroll
      for (int j = 0; j < 8; ++j) acc[a][b][j] += bia;
  }
  if (outF) {
#pragma unroll
    for (int b = 0; b < 4; ++b)
#pragma unroll
      for (int a = 0; a < 2; ++a)
#pragma unroll
        for (int j = 0; j < 8; ++j) {
          const int gm = m0 + wm * 32 + a * 16 + hi8 + j;
          const int gn = n0 + wn * 64 + b * 16 + ln;
          outF[(size_t)gm * N + gn] = acc[a][b][j];
        }
  }
  if (outB) {
#pragma unroll
    for (int b = 0; b < 4; ++b)
#pragma unroll
      for (int a = 0; a < 2; ++a)
#pragma unroll
        for (int j = 0; j < 8; ++j) {
          const int gm = m0 + wm * 32 + a * 16 + hi8 + j;
          const int gn = n0 + wn * 64 + b * 16 + ln;
          outB[(size_t)gm * N + gn] = (__bf16)(acc[a][b][j] * scaleB);
        }
  }
  if (outBT) {  // transposed bf16 [N][M] (for attention V as B-operand)
#pragma unroll
    for (int b = 0; b < 4; ++b)
#pragma unroll
      for (int a = 0; a < 2; ++a)
#pragma unroll
        for (int j = 0; j < 8; ++j) {
          const int gm = m0 + wm * 32 + a * 16 + hi8 + j;
          const int gn = n0 + wn * 64 + b * 16 + ln;
          outBT[(size_t)gn * M + gm] = (__bf16)(acc[a][b][j] * scaleB);
        }
  }
}

// ---------------------------------------------------------------------------
// Attention: 1 wave (32 thr) per block; 16 query rows x one (b,h).
// K/V tiles async double-buffered per wave (no barriers needed).
// Q/K carry dh^-0.25; V read from transposed global vt[1024][3584].
// ---------------------------------------------------------------------------
#define SCST 450
__global__ __launch_bounds__(32) void mha_attn_kernel(
    const __bf16* __restrict__ Q, const __bf16* __restrict__ Kc,
    const __bf16* __restrict__ Vt, __bf16* __restrict__ O) {
  __shared__ float sc[16][SCST];                  // scores/probs (28.8 KB)
  __shared__ __align__(16) __bf16 Ks[2][16][64];  // [buf][key][dh]  (4 KB)
  __shared__ __align__(16) __bf16 Vs[2][64][32];  // [buf][dh][key]  (8 KB)
  __shared__ float rinv[16];

  const int lane = threadIdx.x & 31;
  const int ln = lane & 15, hi8 = (lane >> 4) * 8;
  const int bh = blockIdx.y, b = bh >> 4, h = bh & 15;
  const int q0 = blockIdx.x * 16;

  // Q fragments (dh = 64 -> two K=32 frags), direct from global
  FragB qa[2];
  {
    const __bf16* qp = Q + (size_t)(b * 448 + q0 + ln) * 1024 + h * 64;
#pragma unroll
    for (int t = 0; t < 2; ++t)
#pragma unroll
      for (int j = 0; j < 8; ++j)
        qa[t].u[j] = *(const unsigned*)(qp + t * 32 + fk(j, hi8));
  }

  // async staging: K tile (16 keys x 64 dh, 2 KB = 4 instrs/wave)
  auto stage_k = [&](int key0, int buf) {
#pragma unroll
    for (int i = 0; i < 4; ++i) {
      const int r = i * 4 + (lane >> 3), e = (lane & 7) * 8;
      async_b128(Kc + (size_t)(b * 448 + key0 + r) * 1024 + h * 64 + e,
                 lds_off(&Ks[buf][r][e]));
    }
  };
  // async staging: V tile transposed (64 dh x 32 keys, 4 KB = 8 instrs/wave)
  auto stage_v = [&](int ks, int buf) {
#pragma unroll
    for (int i = 0; i < 8; ++i) {
      const int r = i * 8 + (lane >> 2), e = (lane & 3) * 8;
      async_b128(Vt + (size_t)(h * 64 + r) * 3584 + b * 448 + ks * 32 + e,
                 lds_off(&Vs[buf][r][e]));
    }
  };

  // ---- phase 1: S = q @ k^T + causal mask -> LDS (ASYNCcnt-pipelined) ----
  stage_k(0, 0);
  for (int kt = 0; kt < 28; ++kt) {
    const int buf = kt & 1, key0 = kt * 16;
    if (kt + 1 < 28) { stage_k(key0 + 16, buf ^ 1); WAIT_ASYNC(4); }
    else             { WAIT_ASYNC(0); }
    FragB kf[2];
#pragma unroll
    for (int t = 0; t < 2; ++t)
#pragma unroll
      for (int j = 0; j < 8; ++j)
        kf[t].u[j] = *(const unsigned*)&Ks[buf][ln][t * 32 + fk(j, hi8)];
    v8f s = vzero8();
    s = __builtin_amdgcn_wmma_f32_16x16x32_bf16(false, qa[0].v, false, kf[0].v,
                                                (short)0, s, false, false);
    s = __builtin_amdgcn_wmma_f32_16x16x32_bf16(false, qa[1].v, false, kf[1].v,
                                                (short)0, s, false, false);
#pragma unroll
    for (int j = 0; j < 8; ++j) {
      const int ml = hi8 + j;
      const int kg = key0 + ln;
      sc[ml][kg] = s[j] + (kg > q0 + ml ? -1e9f : 0.f);
    }
  }
  __syncthreads();  // single-wave workgroup: S_NOP (ordering via in-order LDS)

  // ---- phase 2: row softmax; halves per lane-pair, combine via shfl_xor ----
  {
    const int r = lane & 15, c0 = (lane >> 4) * 224;
    float mx = -3.4e38f;
    for (int c = c0; c < c0 + 224; ++c) mx = fmaxf(mx, sc[r][c]);
    mx = fmaxf(mx, __shfl_xor(mx, 16, 32));
    float sum = 0.f;
    for (int c = c0; c < c0 + 224; ++c) {
      float p = __expf(sc[r][c] - mx);
      sc[r][c] = p;
      sum += p;
    }
    sum += __shfl_xor(sum, 16, 32);
    if (lane < 16) rinv[r] = 1.f / sum;
  }
  __syncthreads();

  // ---- phase 3: out = P @ V (ASYNCcnt-pipelined V tiles) ----
  v8f acc[4];
#pragma unroll
  for (int nt = 0; nt < 4; ++nt) acc[nt] = vzero8();

  stage_v(0, 0);
  for (int ks = 0; ks < 14; ++ks) {
    const int buf = ks & 1;
    if (ks + 1 < 14) { stage_v(ks + 1, buf ^ 1); WAIT_ASYNC(8); }
    else             { WAIT_ASYNC(0); }
    FragB pa;  // probs f32 -> bf16 on the fly
#pragma unroll
    for (int j = 0; j < 8; ++j) {
      const int k0 = ks * 32 + fk(j, hi8);
      pa.u[j] = pack2bf(sc[ln][k0], sc[ln][k0 + 1]);
    }
#pragma unroll
    for (int nt = 0; nt < 4; ++nt) {
      FragB vf;
#pragma unroll
      for (int j = 0; j < 8; ++j)
        vf.u[j] = *(const unsigned*)&Vs[buf][nt * 16 + ln][fk(j, hi8)];
      acc[nt] = __builtin_amdgcn_wmma_f32_16x16x32_bf16(
          false, pa.v, false, vf.v, (short)0, acc[nt], false, false);
    }
  }

  // epilogue: 1/rowsum, store bf16 head slice
#pragma unroll
  for (int nt = 0; nt < 4; ++nt)
#pragma unroll
    for (int j = 0; j < 8; ++j) {
      const int ml = hi8 + j;
      const float val = acc[nt][j] * rinv[ml];
      O[(size_t)(b * 448 + q0 + ml) * 1024 + h * 64 + nt * 16 + ln] = (__bf16)val;
    }
}

// ---------------------------------------------------------------------------
// Launcher. Inputs: x, k_cache, v_cache, mask, Wq, bq, Wk, Wv, bv, Wo, bo.
// Output: [out | k_cache | v_cache] (T_new==T -> caches fully overwritten).
// ---------------------------------------------------------------------------
extern "C" void kernel_launch(void* const* d_in, const int* in_sizes, int n_in,
                              void* d_out, int out_size, void* d_ws, size_t ws_size,
                              hipStream_t stream) {
  (void)in_sizes; (void)n_in; (void)out_size; (void)ws_size;
  const float* x  = (const float*)d_in[0];
  const float* Wq = (const float*)d_in[4];
  const float* bq = (const float*)d_in[5];
  const float* Wk = (const float*)d_in[6];
  const float* Wv = (const float*)d_in[7];
  const float* bv = (const float*)d_in[8];
  const float* Wo = (const float*)d_in[9];
  const float* bo = (const float*)d_in[10];
  float* out = (float*)d_out;

  const int    MROWS = 8 * 448;               // 3584
  const size_t MD    = (size_t)MROWS * 1024;  // 3670016 elems per tensor

  // workspace layout (bytes); ~36 MB (xb region reused for attn output)
  char* ws = (char*)d_ws;
  __bf16* xb  = (__bf16*)(ws);                 // 7,340,032
  __bf16* wqT = (__bf16*)(ws + 7340032);       // 2,097,152 each (transposed)
  __bf16* wkT = (__bf16*)(ws + 9437184);
  __bf16* wvT = (__bf16*)(ws + 11534336);
  __bf16* woT = (__bf16*)(ws + 13631488);
  __bf16* qbf = (__bf16*)(ws + 15728640);      // 7,340,032 each
  __bf16* kbf = (__bf16*)(ws + 23068672);
  __bf16* vtg = (__bf16*)(ws + 30408704);      // v, bf16, transposed [1024][3584]
  __bf16* obf = xb;  // xb dead after the three QKV GEMMs

  const float scale = 0.35355339059327373f;  // 64^-0.25, applied to q AND k

  mha_cvt_bf16_kernel<<<(int)((MD + 255) / 256), 256, 0, stream>>>(x, xb, (int)MD);
  dim3 tb(32, 8), tg(32, 32);
  mha_cvt_t_kernel<<<tg, tb, 0, stream>>>(Wq, wqT, 1024, 1024);
  mha_cvt_t_kernel<<<tg, tb, 0, stream>>>(Wk, wkT, 1024, 1024);
  mha_cvt_t_kernel<<<tg, tb, 0, stream>>>(Wv, wvT, 1024, 1024);
  mha_cvt_t_kernel<<<tg, tb, 0, stream>>>(Wo, woT, 1024, 1024);

  dim3 gg(1024 / 128, MROWS / 128);  // (8, 28)
  // q = x@Wq + bq  -> bf16 (scaled)
  mha_gemm_bf16_kernel<<<gg, 256, 0, stream>>>(xb, wqT, bq, nullptr, qbf, nullptr,
                                               scale, MROWS, 1024, 1024);
  // k = x@Wk       -> f32 cache output + bf16 (scaled)
  mha_gemm_bf16_kernel<<<gg, 256, 0, stream>>>(xb, wkT, nullptr, out + MD, kbf,
                                               nullptr, scale, MROWS, 1024, 1024);
  // v = x@Wv + bv  -> f32 cache output + bf16 transposed [D][B*T]
  mha_gemm_bf16_kernel<<<gg, 256, 0, stream>>>(xb, wvT, bv, out + 2 * MD, nullptr,
                                               vtg, 1.0f, MROWS, 1024, 1024);
  // attention per (b,h), 16-query blocks, 1 wave each
  mha_attn_kernel<<<dim3(28, 128), 32, 0, stream>>>(qbf, kbf, vtg, obf);
  // out = attn @ Wo + bo -> f32
  mha_gemm_bf16_kernel<<<gg, 256, 0, stream>>>(obf, woT, bo, out, nullptr, nullptr,
                                               1.0f, MROWS, 1024, 1024);
}
